// HardMaxAttention_28733331210242
// MI455X (gfx1250) — matrix operand order repaired and checked by reading, not compiled, not run on hardware
//
#include <hip/hip_runtime.h>
#include <hip/hip_bf16.h>

typedef __attribute__((ext_vector_type(16))) _Float16 v16h;
typedef __attribute__((ext_vector_type(8)))  _Float16 v8h;
typedef __attribute__((ext_vector_type(8)))  float    v8f;
typedef __attribute__((ext_vector_type(4)))  float    v4f;

constexpr int kB   = 8;
constexpr int kSeq = 2048;   // Q == K
constexpr int kH   = 512;
constexpr int kM   = kB * kSeq;   // 16384 flattened rows

// ---------------------------------------------------------------------------
// Fragment loaders matching CDNA5 WMMA 16x16x32 f16 VGPR layouts (ISA 7.12.2)
// A (16x32, 16-bit): lane m=lane&15 is the row; lanes 0-15 hold K=kb..kb+7 and
// kb+16..kb+23 with kb=0; lanes 16-31 same rows with kb=8.
// B (32x16, 16-bit): lane n=lane&15 is the column; same K order as A
// (kb..kb+7, kb+16..kb+23; kb=0 / 8 per half) -> load from (N,K) storage.
// C/D (16x16 f32): vgpr r, lane -> M = r + 8*(lane>>4), N = lane&15.
// ---------------------------------------------------------------------------

__device__ inline v16h load_a_f16(const _Float16* __restrict__ tile, int ld) {
  int lane = threadIdx.x & 31;
  int m  = lane & 15;
  int kb = (lane >> 4) << 3;
  const _Float16* p = tile + m * ld + kb;
  v8h lo = *(const v8h*)(p);
  v8h hi = *(const v8h*)(p + 16);
  v16h a;
#pragma unroll
  for (int i = 0; i < 8; ++i) { a[i] = lo[i]; a[8 + i] = hi[i]; }
  return a;
}

__device__ inline v16h load_a_f32(const float* __restrict__ tile, int ld) {
  int lane = threadIdx.x & 31;
  int m  = lane & 15;
  int kb = (lane >> 4) << 3;
  const float* p = tile + m * ld + kb;
  v4f x0 = *(const v4f*)(p);
  v4f x1 = *(const v4f*)(p + 4);
  v4f x2 = *(const v4f*)(p + 16);
  v4f x3 = *(const v4f*)(p + 20);
  v16h a;
#pragma unroll
  for (int i = 0; i < 4; ++i) {
    a[i]      = (_Float16)x0[i];
    a[4 + i]  = (_Float16)x1[i];
    a[8 + i]  = (_Float16)x2[i];
    a[12 + i] = (_Float16)x3[i];
  }
  return a;
}

// B fragment from (N,K) row-major f16 storage (i.e. already-transposed weight
// or the K-matrix of Q@K^T where rows are keys).
__device__ inline v16h load_b_nk(const _Float16* __restrict__ tile, int ld) {
  int lane = threadIdx.x & 31;
  int n  = lane & 15;
  int kb = (lane >> 4) << 3;
  const _Float16* p = tile + n * ld + kb;
  v8h lo = *(const v8h*)(p);
  v8h hi = *(const v8h*)(p + 16);
  v16h b;
#pragma unroll
  for (int i = 0; i < 8; ++i) { b[i] = lo[i]; b[8 + i] = hi[i]; }
  return b;
}

#define WMMA_F16(c, a, b) \
  __builtin_amdgcn_wmma_f32_16x16x32_f16(false, (a), false, (b), (short)0, (c), false, false)

// ---------------------------------------------------------------------------
// 0) Convert the four HxH f32 weights into transposed (N,K) f16 copies.
// ---------------------------------------------------------------------------
__global__ void convert_weights_kernel(const float* __restrict__ Wq,
                                       const float* __restrict__ Wk,
                                       const float* __restrict__ Wv,
                                       const float* __restrict__ Wo,
                                       _Float16* __restrict__ Wqt,
                                       _Float16* __restrict__ Wkt,
                                       _Float16* __restrict__ Wvt,
                                       _Float16* __restrict__ Wot) {
  const float* src;
  _Float16* dst;
  switch (blockIdx.z) {
    case 0: src = Wq; dst = Wqt; break;
    case 1: src = Wk; dst = Wkt; break;
    case 2: src = Wv; dst = Wvt; break;
    default: src = Wo; dst = Wot; break;
  }
  int k = blockIdx.x * blockDim.x + threadIdx.x;  // fast dim of dst
  int n = blockIdx.y;
  dst[n * kH + k] = (_Float16)src[k * kH + n];    // Wt[n][k] = W[k][n]
}

// ---------------------------------------------------------------------------
// 1) Projection GEMM: Y(f16, MxH) = X(f32, MxH) @ W + b, W given as Wt (N,K) f16.
//    Block = 8 waves; each wave computes a 16x64 strip (4 accumulators).
// ---------------------------------------------------------------------------
__global__ void __launch_bounds__(256)
proj_gemm_kernel(const float* __restrict__ X,
                 const _Float16* __restrict__ Wt,
                 const float* __restrict__ bias,
                 _Float16* __restrict__ Y) {
  int wave = threadIdx.x >> 5;
  int lane = threadIdx.x & 31;
  int m0 = (blockIdx.x * 8 + wave) * 16;
  int n0 = blockIdx.y * 64;

  v8f acc[4] = {};
  for (int k0 = 0; k0 < kH; k0 += 32) {
    v16h a = load_a_f32(X + m0 * kH + k0, kH);
#pragma unroll
    for (int j = 0; j < 4; ++j) {
      v16h b = load_b_nk(Wt + (n0 + 16 * j) * kH + k0, kH);
      acc[j] = WMMA_F16(acc[j], a, b);
    }
  }

  int hi   = lane >> 4;
  int col0 = lane & 15;
#pragma unroll
  for (int j = 0; j < 4; ++j) {
    int col = n0 + 16 * j + col0;
    float bv = bias[col];
#pragma unroll
    for (int r = 0; r < 8; ++r) {
      int row = m0 + r + 8 * hi;
      Y[row * kH + col] = (_Float16)(acc[j][r] + bv);
    }
  }
}

// ---------------------------------------------------------------------------
// 2) Energy + masked argmax. One wave owns 16 query rows; Q-tile (16x512) kept
//    in 128 VGPRs; streams 16-key tiles of Kp, 16 WMMAs per tile, folds each
//    16x16 energy tile into a per-lane running argmax, then reduces across the
//    16 lanes of each half-wave. Scale 1/sqrt(H) dropped (monotone).
// ---------------------------------------------------------------------------
__global__ void __launch_bounds__(256)
energy_argmax_kernel(const _Float16* __restrict__ Qp,
                     const _Float16* __restrict__ Kp,
                     const unsigned char* __restrict__ mask,
                     int* __restrict__ idx_out) {
  int wave = blockIdx.x * (blockDim.x >> 5) + (threadIdx.x >> 5);
  int lane = threadIdx.x & 31;
  int q0 = wave * 16;          // global flattened (b*Seq + q) row
  int b  = q0 / kSeq;          // tiles never straddle batches (2048 % 16 == 0)

  v16h a[16];
#pragma unroll
  for (int kk = 0; kk < 16; ++kk)
    a[kk] = load_a_f16(Qp + q0 * kH + kk * 32, kH);

  float best[8];
  int   bidx[8];
#pragma unroll
  for (int r = 0; r < 8; ++r) { best[r] = -__builtin_huge_valf(); bidx[r] = 0; }

  int colLane = lane & 15;
  const _Float16* Kb = Kp + b * kSeq * kH;
  const unsigned char* mb = mask + b * kSeq;

  for (int n0 = 0; n0 < kSeq; n0 += 16) {
    v8f c = {};
#pragma unroll
    for (int kk = 0; kk < 16; ++kk) {
      v16h bf = load_b_nk(Kb + n0 * kH + kk * 32, kH);
      c = WMMA_F16(c, a[kk], bf);
    }
    int kcol = n0 + colLane;
    if (!mb[kcol]) {   // masked columns act as -inf: simply never update
#pragma unroll
      for (int r = 0; r < 8; ++r)
        if (c[r] > best[r]) { best[r] = c[r]; bidx[r] = kcol; }  // strict >: first max wins
    }
  }

  // reduce argmax across the 16 lanes of each half-wave (rows r+8*(lane>>4))
#pragma unroll
  for (int off = 1; off < 16; off <<= 1) {
#pragma unroll
    for (int r = 0; r < 8; ++r) {
      float ov = __shfl_xor(best[r], off, 16);
      int   oi = __shfl_xor(bidx[r], off, 16);
      if (ov > best[r] || (ov == best[r] && oi < bidx[r])) {
        best[r] = ov; bidx[r] = oi;
      }
    }
  }

  if (colLane == 0) {
    int hi = lane >> 4;
#pragma unroll
    for (int r = 0; r < 8; ++r)
      idx_out[q0 + r + 8 * hi] = bidx[r];
  }
}

// ---------------------------------------------------------------------------
// 3) score one-hot: zero-fill then scatter 1.0 at argmax.
// ---------------------------------------------------------------------------
__global__ void zero_f32_kernel(float4* __restrict__ p, long long n4) {
  long long i = (long long)blockIdx.x * blockDim.x + threadIdx.x;
  long long stride = (long long)gridDim.x * blockDim.x;
  float4 z = make_float4(0.f, 0.f, 0.f, 0.f);
  for (; i < n4; i += stride) p[i] = z;
}

__global__ void scatter_score_kernel(const int* __restrict__ idx,
                                     float* __restrict__ score) {
  int m = blockIdx.x * blockDim.x + threadIdx.x;   // flattened (b*Seq + q)
  if (m < kM) score[(long long)m * kSeq + idx[m]] = 1.0f;
}

// ---------------------------------------------------------------------------
// 4) Output projection with row gather: res = Vp[idx] @ Wo + bo (f32 out).
// ---------------------------------------------------------------------------
__global__ void __launch_bounds__(256)
out_gemm_kernel(const _Float16* __restrict__ Vp,
                const int* __restrict__ idx,
                const _Float16* __restrict__ Wot,
                const float* __restrict__ bias,
                float* __restrict__ Y) {
  int wave = threadIdx.x >> 5;
  int lane = threadIdx.x & 31;
  int m0 = (blockIdx.x * 8 + wave) * 16;
  int n0 = blockIdx.y * 64;
  int b  = m0 / kSeq;

  // per-lane gathered A-row base (lane&15 = row within tile, both halves)
  int myrow = m0 + (lane & 15);
  const _Float16* arow = Vp + (long long)(b * kSeq + idx[myrow]) * kH;
  int kb = (lane >> 4) << 3;

  v8f acc[4] = {};
  for (int k0 = 0; k0 < kH; k0 += 32) {
    const _Float16* p = arow + k0 + kb;
    v8h lo = *(const v8h*)(p);
    v8h hi = *(const v8h*)(p + 16);
    v16h a;
#pragma unroll
    for (int i = 0; i < 8; ++i) { a[i] = lo[i]; a[8 + i] = hi[i]; }
#pragma unroll
    for (int j = 0; j < 4; ++j) {
      v16h bfrag = load_b_nk(Wot + (n0 + 16 * j) * kH + k0, kH);
      acc[j] = WMMA_F16(acc[j], a, bfrag);
    }
  }

  int hi2  = lane >> 4;
  int col0 = lane & 15;
#pragma unroll
  for (int j = 0; j < 4; ++j) {
    int col = n0 + 16 * j + col0;
    float bv = bias[col];
#pragma unroll
    for (int r = 0; r < 8; ++r) {
      int row = m0 + r + 8 * hi2;
      Y[row * kH + col] = acc[j][r] + bv;
    }
  }
}

// ---------------------------------------------------------------------------
extern "C" void kernel_launch(void* const* d_in, const int* in_sizes, int n_in,
                              void* d_out, int out_size, void* d_ws, size_t ws_size,
                              hipStream_t stream) {
  const float* query = (const float*)d_in[0];
  const float* key_  = (const float*)d_in[1];
  const float* value = (const float*)d_in[2];
  const unsigned char* mask = (const unsigned char*)d_in[3];   // jax bool = 1 byte
  const float* Wq = (const float*)d_in[4];
  const float* bq = (const float*)d_in[5];
  const float* Wk = (const float*)d_in[6];
  const float* bk = (const float*)d_in[7];
  const float* Wv = (const float*)d_in[8];
  const float* bv = (const float*)d_in[9];
  const float* Wo = (const float*)d_in[10];
  const float* bo = (const float*)d_in[11];

  // workspace layout (f16 halves unless noted)
  _Float16* Wqt = (_Float16*)d_ws;                 // H*H
  _Float16* Wkt = Wqt + kH * kH;
  _Float16* Wvt = Wkt + kH * kH;
  _Float16* Wot = Wvt + kH * kH;
  _Float16* Qp  = Wot + kH * kH;                   // M*H each
  _Float16* Kp  = Qp + (long long)kM * kH;
  _Float16* Vp  = Kp + (long long)kM * kH;
  int*      idx = (int*)(Vp + (long long)kM * kH); // M ints

  float* res   = (float*)d_out;                    // (B,Q,H)
  float* score = res + (long long)kM * kH;         // (B,Q,K)

  // 0) weights -> transposed f16
  convert_weights_kernel<<<dim3(kH / 256, kH, 4), 256, 0, stream>>>(
      Wq, Wk, Wv, Wo, Wqt, Wkt, Wvt, Wot);

  // 1) projections (M=16384, block tile 128x64)
  dim3 gproj(kM / 128, kH / 64);
  proj_gemm_kernel<<<gproj, 256, 0, stream>>>(query, Wqt, bq, Qp);
  proj_gemm_kernel<<<gproj, 256, 0, stream>>>(key_,  Wkt, bk, Kp);
  proj_gemm_kernel<<<gproj, 256, 0, stream>>>(value, Wvt, bv, Vp);

  // 2) energy + argmax (1024 waves, 16 query rows each)
  energy_argmax_kernel<<<kM / 16 / 8, 256, 0, stream>>>(Qp, Kp, mask, idx);

  // 3) one-hot score
  long long n4 = (long long)kM * kSeq / 4;
  zero_f32_kernel<<<8192, 256, 0, stream>>>((float4*)score, n4);
  scatter_score_kernel<<<kM / 256, 256, 0, stream>>>(idx, score);

  // 4) gathered output projection
  out_gemm_kernel<<<gproj, 256, 0, stream>>>(Vp, idx, Wot, bo, res);
}
